// RVMG_20040317403816
// MI455X (gfx1250) — matrix operand order repaired and checked by reading, not compiled, not run on hardware
//
#include <hip/hip_runtime.h>
#include <hip/hip_bf16.h>
#include <math.h>
#include <cstddef>

// Problem dims (match reference)
#define BB    2
#define FFR   8
#define HHR   64
#define WWR   64
#define CCH   64
#define DIM   96
#define DST   16
#define DTRK  4
#define DEP   4
#define HWSZ  (HHR*WWR)        // 4096
#define LVSEQ (FFR*HWSZ)       // 32768
#define MROWS (BB*LVSEQ)       // 65536
#define CHUNK 512
#define NCH   (LVSEQ/CHUNK)    // 64 chunks per batch

typedef __attribute__((ext_vector_type(16))) __bf16       v16bf;
typedef __attribute__((ext_vector_type(8)))  float        v8f;
typedef __attribute__((ext_vector_type(4)))  unsigned int u32x4;
typedef __attribute__((ext_vector_type(4)))  int          v4i;
typedef unsigned short ushort;

// address-space-qualified int4 for the async-to-LDS builtin
typedef __attribute__((address_space(1))) v4i gv4i;   // global
typedef __attribute__((address_space(3))) v4i lv4i;   // LDS

union Frag { v16bf v; u32x4 q[2]; ushort s[16]; };

// CDNA5 async global->LDS path (ASYNCcnt); guarded per-toolchain.
#if defined(__has_builtin)
#  if __has_builtin(__builtin_amdgcn_global_load_async_to_lds_b128) && \
      __has_builtin(__builtin_amdgcn_s_wait_asynccnt)
#    define USE_ASYNC_LDS 1
#  endif
#endif

__device__ __forceinline__ ushort f2bf(float f) {
  unsigned u = __builtin_bit_cast(unsigned, f);
  u += 0x7FFFu + ((u >> 16) & 1u);          // round-to-nearest-even
  return (ushort)(u >> 16);
}
__device__ __forceinline__ float siluf(float x) { return x / (1.f + __expf(-x)); }
__device__ __forceinline__ float geluf(float x) {
  float x3 = x * x * x;
  return 0.5f * x * (1.f + tanhf(0.7978845608028654f * (x + 0.044715f * x3)));
}

// ---------------- cond projection: cb[b, 0:128] = cdp[b] @ w (128x128) -------
__global__ void k_cond(const float* __restrict__ cdp, const float* __restrict__ w,
                       float* __restrict__ cb) {
  int t = threadIdx.x;
  int b = t >> 7, j = t & 127;
  float acc = 0.f;
  #pragma unroll 8
  for (int k = 0; k < 128; ++k) acc += cdp[b * 128 + k] * w[k * 128 + j];
  cb[t] = acc;
}

// ---------------- weight swizzle into B-fragment order -----------------------
// out[((nt*NKc + kk)*32 + lane)*16 + e] = W[k, n], n = nt*16+lane%16,
// k = kk*32 + (lane/16)*16 + e ; zero-pad n >= N.
__global__ void k_wswz(const float* __restrict__ W, ushort* __restrict__ out,
                       int N, int K, int total) {
  int idx = blockIdx.x * 256 + threadIdx.x;
  if (idx >= total) return;
  int e    = idx & 15;
  int lane = (idx >> 4) & 31;
  int NKc  = K >> 5;
  int kk   = (idx >> 9) % NKc;
  int nt   = idx / (NKc << 9);
  int n    = nt * 16 + (lane & 15);
  int k    = kk * 32 + (lane >> 4) * 16 + e;
  out[idx] = (n < N) ? f2bf(W[(size_t)k * N + n]) : (ushort)0;
}

// conv3d weights (O=64,I=64,3,3,3) -> [tap][kk][ntile][lane][16]
__global__ void k_wswz3(const float* __restrict__ Wc, ushort* __restrict__ out) {
  int idx = blockIdx.x * 256 + threadIdx.x;     // 27*2*4*512 = 110592
  if (idx >= 27 * 4096) return;
  int e    = idx & 15;
  int lane = (idx >> 4) & 31;
  int nt   = (idx >> 9) & 3;
  int kk   = (idx >> 11) & 1;
  int tap  = idx >> 12;
  int n    = nt * 16 + (lane & 15);
  int i    = kk * 32 + (lane >> 4) * 16 + e;
  out[idx] = f2bf(Wc[((size_t)n * CCH + i) * 27 + tap]);
}

// ---------------- VDIM: LayerNorm(C=64)*g+b then *c1+c2 -> bf16 --------------
__global__ void k_vdim(const float* __restrict__ X, const float* __restrict__ cb,
                       const float* __restrict__ g, const float* __restrict__ bta,
                       ushort* __restrict__ out) {
  int row  = blockIdx.x * 8 + (threadIdx.x >> 5);
  int lane = threadIdx.x & 31;
  const float* xr = X + (size_t)row * CCH;
  float x0 = xr[lane], x1 = xr[lane + 32];
  float sum = x0 + x1;
  #pragma unroll
  for (int off = 1; off < 32; off <<= 1) sum += __shfl_xor(sum, off, 32);
  float mu = sum * (1.f / 64.f);
  float d0 = x0 - mu, d1 = x1 - mu;
  float vs = d0 * d0 + d1 * d1;
  #pragma unroll
  for (int off = 1; off < 32; off <<= 1) vs += __shfl_xor(vs, off, 32);
  float inv = rsqrtf(vs * (1.f / 64.f) + 1e-6f);
  int bidx = row >> 15;
  const float* c1 = cb + bidx * 128;
  const float* c2 = c1 + 64;
  out[(size_t)row * CCH + lane] =
      f2bf((d0 * inv * g[lane] + bta[lane]) * c1[lane] + c2[lane]);
  out[(size_t)row * CCH + lane + 32] =
      f2bf((d1 * inv * g[lane + 32] + bta[lane + 32]) * c1[lane + 32] + c2[lane + 32]);
}

// ---------------- WMMA GEMM: 32 rows/wave, B panel staged in LDS -------------
// out = act(A@W + bias) [+ resid*skip]; outF (f32) and/or outB (bf16)
template <int NK>
__global__ void k_gemm(const ushort* __restrict__ A, const ushort* __restrict__ Wsw,
                       const float* __restrict__ bias,
                       const float* resid, const float* __restrict__ skip,
                       float* outF, ushort* outB, int N, int act) {
  __shared__ ushort sW[24 * 512];            // up to 24 fragment groups (24 KB)
  const int tid  = threadIdx.x;
  const int wave = tid >> 5;
  const int lane = tid & 31;
  const int hh   = lane >> 4;
  const int lm   = lane & 15;
  const int m0   = blockIdx.x * 128 + wave * 32;
  const int K    = NK * 32;
  const int NT   = (N + 15) >> 4;
  const int G    = NT * NK;                  // groups of 512 ushort (1 KB)

  // stage whole swizzled weight panel into LDS once per block
#ifdef USE_ASYNC_LDS
  for (int c = tid; c < G * 64; c += 128) {  // 16-byte chunks
    __builtin_amdgcn_global_load_async_to_lds_b128(
        (gv4i*)(Wsw + (size_t)c * 8), (lv4i*)(sW + (size_t)c * 8), 0, 0);
  }
  __builtin_amdgcn_s_wait_asynccnt(0);
  __syncthreads();
#else
  for (int c = tid; c < G * 64; c += 128)
    *(u32x4*)(sW + (size_t)c * 8) = *(const u32x4*)(Wsw + (size_t)c * 8);
  __syncthreads();
#endif

  Frag a0[NK], a1[NK];
  const ushort* ar0 = A + (size_t)(m0 + lm) * K + 16 * hh;
  const ushort* ar1 = ar0 + (size_t)16 * K;
  #pragma unroll
  for (int kk = 0; kk < NK; ++kk) {          // two b128 loads per fragment
    a0[kk].q[0] = *(const u32x4*)(ar0 + kk * 32);
    a0[kk].q[1] = *(const u32x4*)(ar0 + kk * 32 + 8);
    a1[kk].q[0] = *(const u32x4*)(ar1 + kk * 32);
    a1[kk].q[1] = *(const u32x4*)(ar1 + kk * 32 + 8);
  }

  for (int nt = 0; nt < NT; ++nt) {
    int  n  = nt * 16 + lm;
    bool nv = (n < N);
    v8f acc0, acc1;
    float bv = (bias && nv) ? bias[n] : 0.f;
    #pragma unroll
    for (int r = 0; r < 8; ++r) { acc0[r] = bv; acc1[r] = bv; }

    #pragma unroll
    for (int kk = 0; kk < NK; ++kk) {
      Frag bf;
      const ushort* wp = sW + (((size_t)nt * NK + kk) << 9) + lane * 16;
      bf.q[0] = *(const u32x4*)(wp);         // ds_load_b128 (broadcast across waves)
      bf.q[1] = *(const u32x4*)(wp + 8);
      acc0 = __builtin_amdgcn_wmma_f32_16x16x32_bf16(
          false, a0[kk].v, false, bf.v, (short)0, acc0, false, false);
      acc1 = __builtin_amdgcn_wmma_f32_16x16x32_bf16(
          false, a1[kk].v, false, bf.v, (short)0, acc1, false, false);
    }
    if (nv) {
      #pragma unroll
      for (int r = 0; r < 8; ++r) {
        int m_a = m0 + hh * 8 + r;
        int m_b = m_a + 16;
        float va = acc0[r], vb = acc1[r];
        if (act == 2) { va = geluf(va); vb = geluf(vb); }
        if (resid) {
          va += resid[(size_t)m_a * N + n] * skip[n];
          vb += resid[(size_t)m_b * N + n] * skip[n];
        }
        if (outF) { outF[(size_t)m_a * N + n] = va; outF[(size_t)m_b * N + n] = vb; }
        if (outB) { outB[(size_t)m_a * N + n] = f2bf(va); outB[(size_t)m_b * N + n] = f2bf(vb); }
      }
    }
  }
}

// ---------------- depthwise causal conv1d (width 4) + SiLU -------------------
__global__ void k_conv1d(const float* __restrict__ xz, const float* __restrict__ cw,
                         const float* __restrict__ cbi,
                         ushort* __restrict__ xmbf, float* __restrict__ xmf) {
  size_t idx = (size_t)blockIdx.x * blockDim.x + threadIdx.x;
  if (idx >= (size_t)MROWS * DIM) return;
  int    d = (int)(idx % DIM);
  size_t m = idx / DIM;
  int    l = (int)(m & (LVSEQ - 1));
  float acc = cbi[d];
  #pragma unroll
  for (int j = 0; j < 4; ++j) {
    int lt = l - 3 + j;
    if (lt >= 0) acc += xz[(m - 3 + j) * (2 * DIM) + d] * cw[d * 4 + j];
  }
  float v = siluf(acc);
  xmf[idx]  = v;
  xmbf[idx] = f2bf(v);
}

// ---------------- dt = softplus(proj[:, :4] @ dtw + dtb) ---------------------
__global__ void k_dtsp(const float* __restrict__ proj, const float* __restrict__ dtw,
                       const float* __restrict__ dtb, float* __restrict__ dt) {
  size_t idx = (size_t)blockIdx.x * blockDim.x + threadIdx.x;
  if (idx >= (size_t)MROWS * DIM) return;
  int    d = (int)(idx % DIM);
  size_t m = idx / DIM;
  const float* pr = proj + m * 36;
  float acc = dtb[d];
  #pragma unroll
  for (int j = 0; j < DTRK; ++j) acc += pr[j] * dtw[j * DIM + d];
  dt[idx] = (acc > 20.f) ? acc : log1pf(__expf(acc));
}

// ---------------- chunked parallel selective scan ----------------------------
// pass1: per-chunk local scan from h=0; emit P = prod(dA), Hend.
__global__ void k_scan1(const float* __restrict__ u, const float* __restrict__ dt,
                        const float* __restrict__ proj, const float* __restrict__ Alog,
                        float* __restrict__ Pst, float* __restrict__ Hend) {
  int blk = blockIdx.x;                    // ((b*NCH + c)*48 + dp)
  int dp  = blk % (DIM / 2);
  int bc  = blk / (DIM / 2);
  int c   = bc % NCH, b = bc / NCH;
  int lane = threadIdx.x & 31;
  int hh = lane >> 4, s = lane & 15;
  int d  = dp * 2 + hh;
  float Ad = -__expf(Alog[d * DST + s]);
  float h = 0.f, P = 1.f;
  const size_t base = (size_t)b * LVSEQ + (size_t)c * CHUNK;
  for (int t = 0; t < CHUNK; ++t) {
    size_t m = base + t;
    if ((t & 15) == 0) {
      __builtin_prefetch(proj + (m + 16) * 36, 0, 1);
      __builtin_prefetch(dt + (m + 16) * DIM + d, 0, 1);
      __builtin_prefetch(u + (m + 16) * DIM + d, 0, 1);
    }
    float dtv = dt[m * DIM + d];
    float uv  = u[m * DIM + d];
    float Bv  = proj[m * 36 + 4 + s];
    float dA  = __expf(dtv * Ad);
    h = dA * h + dtv * uv * Bv;
    P *= dA;
  }
  size_t o = (((size_t)b * NCH + c) * DIM + d) * DST + s;
  Pst[o] = P;
  Hend[o] = h;
}

// pass2: sequential combine over 64 chunk summaries -> carry-in per chunk.
__global__ void k_scan2(const float* __restrict__ Pst, const float* __restrict__ Hend,
                        float* __restrict__ Carr) {
  int blk = blockIdx.x;                    // b*48 + dp
  int dp  = blk % (DIM / 2);
  int b   = blk / (DIM / 2);
  int lane = threadIdx.x & 31;
  int hh = lane >> 4, s = lane & 15;
  int d  = dp * 2 + hh;
  float carry = 0.f;
  for (int c = 0; c < NCH; ++c) {
    size_t o = (((size_t)b * NCH + c) * DIM + d) * DST + s;
    Carr[o] = carry;
    carry = Pst[o] * carry + Hend[o];
  }
}

// pass3: rerun each chunk with carry-in; emit y = sum_s h*C + u*D.
__global__ void k_scan3(const float* __restrict__ u, const float* __restrict__ dt,
                        const float* __restrict__ proj, const float* __restrict__ Alog,
                        const float* __restrict__ Dp, const float* __restrict__ Carr,
                        float* __restrict__ y) {
  int blk = blockIdx.x;
  int dp  = blk % (DIM / 2);
  int bc  = blk / (DIM / 2);
  int c   = bc % NCH, b = bc / NCH;
  int lane = threadIdx.x & 31;
  int hh = lane >> 4, s = lane & 15;
  int d  = dp * 2 + hh;
  float Ad = -__expf(Alog[d * DST + s]);
  float Dd = Dp[d];
  float h  = Carr[(((size_t)b * NCH + c) * DIM + d) * DST + s];
  const size_t base = (size_t)b * LVSEQ + (size_t)c * CHUNK;
  for (int t = 0; t < CHUNK; ++t) {
    size_t m = base + t;
    if ((t & 15) == 0) {
      __builtin_prefetch(proj + (m + 16) * 36, 0, 1);
      __builtin_prefetch(dt + (m + 16) * DIM + d, 0, 1);
      __builtin_prefetch(u + (m + 16) * DIM + d, 0, 1);
    }
    float dtv = dt[m * DIM + d];
    float uv  = u[m * DIM + d];
    float Bv  = proj[m * 36 + 4 + s];
    float Cv  = proj[m * 36 + 20 + s];
    float dA  = __expf(dtv * Ad);
    h = dA * h + dtv * uv * Bv;
    float p = h * Cv;
    p += __shfl_xor(p, 1, 32);
    p += __shfl_xor(p, 2, 32);
    p += __shfl_xor(p, 4, 32);
    p += __shfl_xor(p, 8, 32);
    if (s == 0) y[m * DIM + d] = p + uv * Dd;
  }
}

// ---------------- y_gated = y * silu(z) -> bf16 ------------------------------
__global__ void k_gate(const float* __restrict__ y, const float* __restrict__ xz,
                       ushort* __restrict__ out) {
  size_t idx = (size_t)blockIdx.x * blockDim.x + threadIdx.x;
  if (idx >= (size_t)MROWS * DIM) return;
  int    d = (int)(idx % DIM);
  size_t m = idx / DIM;
  out[idx] = f2bf(y[idx] * siluf(xz[m * (2 * DIM) + DIM + d]));
}

// ---------------- Conv3D 3x3x3 (64->64) as 27-tap WMMA GEMM ------------------
__global__ void k_conv3d(const ushort* __restrict__ Xbf, const ushort* __restrict__ Wsw,
                         const float* __restrict__ bias, const float* __restrict__ xblk,
                         float* __restrict__ out) {
  int gx = blockIdx.x;                 // (b, f, h, wtile): 2*8*64*4 = 4096
  int wt = gx & 3;
  int h_ = (gx >> 2) & 63;
  int f_ = (gx >> 8) & 7;
  int b_ = gx >> 11;
  int wave = threadIdx.x >> 5;
  int lane = threadIdx.x & 31;
  int hh = lane >> 4, lm = lane & 15;
  int n = wave * 16 + lm;

  v8f acc;
  float bv = bias[n];
  #pragma unroll
  for (int r = 0; r < 8; ++r) acc[r] = bv;

  for (int kd = 0; kd < 3; ++kd) {
    int f2 = f_ + kd - 1;
    if ((unsigned)f2 >= (unsigned)FFR) continue;
    for (int kh = 0; kh < 3; ++kh) {
      int h2 = h_ + kh - 1;
      if ((unsigned)h2 >= (unsigned)HHR) continue;
      const ushort* xrow =
          Xbf + ((size_t)(b_ * FFR + f2) * HWSZ + (size_t)h2 * WWR) * CCH;
      for (int kw = 0; kw < 3; ++kw) {
        int  w2  = wt * 16 + lm + kw - 1;
        bool rv  = (unsigned)w2 < (unsigned)WWR;
        int  tap = kd * 9 + kh * 3 + kw;
        #pragma unroll
        for (int kk = 0; kk < 2; ++kk) {
          Frag af;
          if (rv) {
            const ushort* p = xrow + (size_t)w2 * CCH + kk * 32 + 16 * hh;
            af.q[0] = *(const u32x4*)(p);
            af.q[1] = *(const u32x4*)(p + 8);
          } else {
            af.q[0] = (u32x4)(0u); af.q[1] = (u32x4)(0u);
          }
          Frag bf;
          const ushort* wp = Wsw + (((size_t)(tap * 2 + kk) * 4 + wave) << 9) + lane * 16;
          bf.q[0] = *(const u32x4*)(wp);
          bf.q[1] = *(const u32x4*)(wp + 8);
          acc = __builtin_amdgcn_wmma_f32_16x16x32_bf16(
              false, af.v, false, bf.v, (short)0, acc, false, false);
        }
      }
    }
  }
  size_t rowc = (size_t)(b_ * FFR + f_) * HWSZ + (size_t)h_ * WWR;
  #pragma unroll
  for (int r = 0; r < 8; ++r) {
    int    wo  = wt * 16 + hh * 8 + r;
    size_t idx = (rowc + wo) * CCH + n;
    out[idx] = acc[r] + xblk[idx];
  }
}

// ---------------- host orchestration ----------------------------------------
extern "C" void kernel_launch(void* const* d_in, const int* in_sizes, int n_in,
                              void* d_out, int out_size, void* d_ws, size_t ws_size,
                              hipStream_t stream) {
  (void)in_sizes; (void)n_in; (void)out_size; (void)ws_size;
  const float* x_in  = (const float*)d_in[0];
  const float* cdp   = (const float*)d_in[1];
  const float* v1w   = (const float*)d_in[2];
  const float* v1g   = (const float*)d_in[3];
  const float* v1b   = (const float*)d_in[4];
  const float* v2w   = (const float*)d_in[5];
  const float* v2g   = (const float*)d_in[6];
  const float* v2b   = (const float*)d_in[7];
  const float* skip1 = (const float*)d_in[8];
  const float* skip2 = (const float*)d_in[9];
  const float* mw1   = (const float*)d_in[10];
  const float* mb1   = (const float*)d_in[11];
  const float* mw2   = (const float*)d_in[12];
  const float* mb2   = (const float*)d_in[13];
  const float* inpj  = (const float*)d_in[14];
  const float* c1w   = (const float*)d_in[15];
  const float* c1b   = (const float*)d_in[16];
  const float* xpj   = (const float*)d_in[17];
  const float* dtw   = (const float*)d_in[18];
  const float* dtb   = (const float*)d_in[19];
  const float* alog  = (const float*)d_in[20];
  const float* Dpp   = (const float*)d_in[21];
  const float* outpj = (const float*)d_in[22];
  const float* c3w   = (const float*)d_in[23];
  const float* c3b   = (const float*)d_in[24];

  constexpr size_t NX   = (size_t)MROWS * CCH;
  constexpr size_t NXZ  = (size_t)MROWS * 192;
  constexpr size_t N96  = (size_t)MROWS * DIM;
  constexpr size_t NPRJ = (size_t)MROWS * 36;
  constexpr size_t NSUM = (size_t)BB * NCH * DIM * DST;   // 196,608

  char* base = (char*)d_ws;
  auto alloc = [&](size_t bytes) {
    char* p = base;
    base += (bytes + 255) & ~(size_t)255;
    return p;
  };
  float*  X    = (float*)alloc(NX * 4);
  float*  XZ   = (float*)alloc(NXZ * 4);
  float*  XMf  = (float*)alloc(N96 * 4);
  float*  PROJ = (float*)alloc(NPRJ * 4);
  float*  DT   = (float*)alloc(N96 * 4);
  float*  Y    = (float*)alloc(N96 * 4);
  float*  CB   = (float*)alloc(256 * 4);
  float*  Pst  = (float*)alloc(NSUM * 4);
  float*  Hend = (float*)alloc(NSUM * 4);
  float*  Carr = (float*)alloc(NSUM * 4);
  ushort* Xbf  = (ushort*)alloc(NX * 2);
  ushort* Hbbf = (ushort*)alloc(NX * 2);
  ushort* XMbf = (ushort*)alloc(N96 * 2);
  ushort* YGbf = (ushort*)alloc(N96 * 2);
  ushort* HIDbf= (ushort*)alloc(N96 * 2);
  ushort* WSWZ = (ushort*)alloc(12 * 2 * 512 * 2);        // max NT*NK*512
  ushort* WSW3 = (ushort*)alloc(27 * 4096 * 2);

  (void)hipMemcpyAsync(X, x_in, NX * sizeof(float), hipMemcpyDeviceToDevice, stream);

  const int GE = MROWS * DIM / 256;
  auto swz = [&](const float* W, int N, int K) {
    int total = ((N + 15) / 16) * (K / 32) * 512;
    k_wswz<<<(total + 255) / 256, 256, 0, stream>>>(W, WSWZ, N, K, total);
  };

  for (int i = 0; i < DEP; ++i) {
    // RVMB: VDIM -> Mamba3D, residual x*skip1 + m
    k_cond<<<1, 256, 0, stream>>>(cdp, v1w + (size_t)i * 128 * 128, CB);
    k_vdim<<<MROWS / 8, 256, 0, stream>>>(X, CB, v1g + i * 64, v1b + i * 64, Hbbf);
    swz(inpj + (size_t)i * 64 * 192, 192, 64);
    k_gemm<2><<<MROWS / 128, 128, 0, stream>>>(Hbbf, WSWZ, nullptr, nullptr, nullptr,
                                               XZ, nullptr, 192, 0);
    k_conv1d<<<GE, 256, 0, stream>>>(XZ, c1w + i * 96 * 4, c1b + i * 96, XMbf, XMf);
    swz(xpj + (size_t)i * 96 * 36, 36, 96);
    k_gemm<3><<<MROWS / 128, 128, 0, stream>>>(XMbf, WSWZ, nullptr, nullptr, nullptr,
                                               PROJ, nullptr, 36, 0);
    k_dtsp<<<GE, 256, 0, stream>>>(PROJ, dtw + i * 4 * 96, dtb + i * 96, DT);
    k_scan1<<<BB * NCH * (DIM / 2), 32, 0, stream>>>(XMf, DT, PROJ,
                                                     alog + i * 96 * 16, Pst, Hend);
    k_scan2<<<BB * (DIM / 2), 32, 0, stream>>>(Pst, Hend, Carr);
    k_scan3<<<BB * NCH * (DIM / 2), 32, 0, stream>>>(XMf, DT, PROJ,
                                                     alog + i * 96 * 16, Dpp + i * 96,
                                                     Carr, Y);
    k_gate<<<GE, 256, 0, stream>>>(Y, XZ, YGbf);
    swz(outpj + (size_t)i * 96 * 64, 64, 96);
    k_gemm<3><<<MROWS / 128, 128, 0, stream>>>(YGbf, WSWZ, nullptr, X, skip1 + i * 64,
                                               X, Xbf, 64, 0);
    // VDIM -> MLP, residual x*skip2 + mlp
    k_cond<<<1, 256, 0, stream>>>(cdp, v2w + (size_t)i * 128 * 128, CB);
    k_vdim<<<MROWS / 8, 256, 0, stream>>>(X, CB, v2g + i * 64, v2b + i * 64, Hbbf);
    swz(mw1 + (size_t)i * 64 * 96, 96, 64);
    k_gemm<2><<<MROWS / 128, 128, 0, stream>>>(Hbbf, WSWZ, mb1 + i * 96, nullptr,
                                               nullptr, nullptr, HIDbf, 96, 2);
    swz(mw2 + (size_t)i * 96 * 64, 64, 96);
    k_gemm<3><<<MROWS / 128, 128, 0, stream>>>(HIDbf, WSWZ, mb2 + i * 64, X,
                                               skip2 + i * 64, X, Xbf, 64, 0);
  }
  // residual Conv3D into d_out (= conv(x_final) + bias + x_blk)
  k_wswz3<<<(27 * 4096 + 255) / 256, 256, 0, stream>>>(c3w, WSW3);
  k_conv3d<<<BB * FFR * HHR * (WWR / 16), 128, 0, stream>>>(Xbf, WSW3, c3b, x_in,
                                                            (float*)d_out);
}
